// AttentionLayer_63840393888115
// MI455X (gfx1250) — matrix operand order repaired and checked by reading, not compiled, not run on hardware
//
#include <hip/hip_runtime.h>

// ---------------------------------------------------------------------------
// Types / WMMA helpers (CDNA5 / gfx1250, wave32)
// ---------------------------------------------------------------------------
typedef __attribute__((ext_vector_type(16))) __bf16 v16bf;
typedef __attribute__((ext_vector_type(8)))  float  v8f;

union Frag16 { v16bf v; uint4 q[2]; };

__device__ __forceinline__ v8f wmma_bf16(const v16bf& a, const v16bf& b, const v8f& c) {
    // D = A(16x32 bf16) x B(32x16 bf16) + C(16x16 f32)
    return __builtin_amdgcn_wmma_f32_16x16x32_bf16(
        /*neg_a=*/false, a, /*neg_b=*/false, b,
        /*c_mod=*/(short)0, c, /*reuse_a=*/false, /*reuse_b=*/false);
}

// Problem constants
constexpr int Bb  = 2;
constexpr int Ss  = 2048;
constexpr int Dd  = 1024;
constexpr int Hh  = 16;
constexpr int HD  = 64;
constexpr int Mtot = Bb * Ss;   // 4096
constexpr int Ktot = Dd;        // 1024

// ---------------------------------------------------------------------------
// f32 -> bf16 conversion (one-time; halves HBM traffic for all GEMM operands)
// ---------------------------------------------------------------------------
__global__ __launch_bounds__(256) void cvt_f32_bf16(const float* __restrict__ src,
                                                    __bf16* __restrict__ dst) {
    size_t i = ((size_t)blockIdx.x * 256 + threadIdx.x) * 4;
    float4 f = *(const float4*)(src + i);
    dst[i + 0] = (__bf16)f.x;
    dst[i + 1] = (__bf16)f.y;
    dst[i + 2] = (__bf16)f.z;
    dst[i + 3] = (__bf16)f.w;
}

// ---------------------------------------------------------------------------
// Tiled bf16 GEMM: C[4096 x NTOT] = A[4096 x 1024] * W[1024 x NTOT] + bias
// Block tile 128x128, 8 waves, each wave 32x64 (2x4 WMMA 16x16 tiles), BK=32.
// Software-pipelined: next tile's global loads issue before this tile's WMMAs.
// MODE 0: QKV epilogue -> Q pre-scaled by 1/sqrt(HD); scatter Q[B,H,S,HD],
//         K[B,H,S,HD], Vt[B,H,HD,S] (bf16)
// MODE 1: plain f32 output (final projection)
// ---------------------------------------------------------------------------
template <int MODE, int NTOT>
__global__ __launch_bounds__(256) void gemm_bf16(
    const __bf16* __restrict__ A,     // [4096][1024] row-major bf16
    const __bf16* __restrict__ W,     // [1024][NTOT] row-major bf16
    const float*  __restrict__ bias,  // [NTOT]
    __bf16* __restrict__ Qb, __bf16* __restrict__ Kb, __bf16* __restrict__ Vt,
    float* __restrict__ Out)
{
    __shared__ __bf16 As[128][40];  // [m][k]  (stride 40 -> conflict-free b128 reads)
    __shared__ __bf16 Bs[128][40];  // [n][k]  (W tile stored transposed)

    const int tid  = threadIdx.x;
    const int lane = tid & 31;
    const int wid  = tid >> 5;
    const int lg   = lane >> 4;   // lane half (K-interleave select)
    const int ln   = lane & 15;

    const int m0 = blockIdx.y * 128;
    const int n0 = blockIdx.x * 128;
    const int wm = (wid & 3) * 32;   // wave row offset in block tile
    const int wn = (wid >> 2) * 64;  // wave col offset in block tile

    v8f acc[2][4];
    v8f zero = {};
#pragma unroll
    for (int mi = 0; mi < 2; ++mi)
#pragma unroll
        for (int ni = 0; ni < 4; ++ni) acc[mi][ni] = zero;

    // global->LDS assignments
    const int arow = tid >> 1;            // 0..127
    const int ahalf = (tid & 1) * 16;     // 0 / 16
    const int bk = tid >> 3;              // 0..31
    const int bseg = (tid & 7) * 16;      // 0..112

    const __bf16* ag = A + (size_t)(m0 + arow) * Ktot + ahalf;
    const __bf16* wg = W + (size_t)bk * NTOT + n0 + bseg;

    // prologue: stage first K-tile
    uint4 a0 = *(const uint4*)ag;
    uint4 a1 = *(const uint4*)(ag + 8);
    Frag16 wv;
    wv.q[0] = *(const uint4*)wg;
    wv.q[1] = *(const uint4*)(wg + 8);

    for (int k0 = 0; k0 < Ktot; k0 += 32) {
        __syncthreads();  // previous iteration's LDS reads done
        *(uint4*)&As[arow][ahalf]     = a0;
        *(uint4*)&As[arow][ahalf + 8] = a1;
#pragma unroll
        for (int j = 0; j < 16; ++j) Bs[bseg + j][bk] = wv.v[j];  // transpose into LDS
        __syncthreads();

        // issue next tile's global loads now; they overlap the WMMAs below
        if (k0 + 32 < Ktot) {
            const __bf16* agn = ag + (k0 + 32);
            const __bf16* wgn = wg + (size_t)(k0 + 32) * NTOT;
            a0 = *(const uint4*)agn;
            a1 = *(const uint4*)(agn + 8);
            wv.q[0] = *(const uint4*)wgn;
            wv.q[1] = *(const uint4*)(wgn + 8);
            __builtin_prefetch(agn + 32, 0, 1);               // global_prefetch_b8
            __builtin_prefetch(wgn + (size_t)32 * NTOT, 0, 1);
        }

        // A fragments: lane ln holds row; half lg selects K 0-15 / 16-31
        Frag16 af[2];
#pragma unroll
        for (int mi = 0; mi < 2; ++mi) {
            const __bf16* p = &As[wm + mi * 16 + ln][lg * 16];
            af[mi].q[0] = *(const uint4*)p;
            af[mi].q[1] = *(const uint4*)(p + 8);
        }
        // B fragments: lane ln holds column n; half lg selects K 0-15 / 16-31
        Frag16 bf[4];
#pragma unroll
        for (int ni = 0; ni < 4; ++ni) {
            const __bf16* p = &Bs[wn + ni * 16 + ln][lg * 16];
            bf[ni].q[0] = *(const uint4*)p;
            bf[ni].q[1] = *(const uint4*)(p + 8);
        }
#pragma unroll
        for (int mi = 0; mi < 2; ++mi)
#pragma unroll
            for (int ni = 0; ni < 4; ++ni)
                acc[mi][ni] = wmma_bf16(af[mi].v, bf[ni].v, acc[mi][ni]);
    }

    // epilogue: C element (r, lane) -> row r + 8*lg, col ln within the 16x16 tile
#pragma unroll
    for (int mi = 0; mi < 2; ++mi) {
#pragma unroll
        for (int ni = 0; ni < 4; ++ni) {
#pragma unroll
            for (int r = 0; r < 8; ++r) {
                const int gm = m0 + wm + mi * 16 + r + lg * 8;
                const int gn = n0 + wn + ni * 16 + ln;
                float v = acc[mi][ni][r] + bias[gn];
                if (MODE == 1) {
                    Out[(size_t)gm * NTOT + gn] = v;
                } else {
                    const int sel = gn >> 10;       // 0=Q 1=K 2=V
                    const int d   = gn & 1023;
                    const int h   = d >> 6;
                    const int hd  = d & 63;
                    const int b   = gm >> 11;
                    const int s   = gm & 2047;
                    const size_t bh = (size_t)(b * Hh + h);
                    if (sel == 0) {
                        v *= 0.125f;  // fold 1/sqrt(HD) into Q once
                        Qb[(bh * Ss + s) * HD + hd] = (__bf16)v;
                    } else if (sel == 1) {
                        Kb[(bh * Ss + s) * HD + hd] = (__bf16)v;
                    } else {
                        Vt[(bh * HD + hd) * Ss + s] = (__bf16)v;  // V transposed
                    }
                }
            }
        }
    }
}

// ---------------------------------------------------------------------------
// Online-softmax stats for one 16x32 score tile (two 16x16 f32 accumulators).
// Produces per-row rescale factor alpha; does NOT touch the output accumulators
// (caller applies alpha on a single code path so regalloc keeps them in place).
// MASKED=false: causally interior tile -> no compare/select work at all.
// c = kbase + ln - (q0 + prow): element r is masked iff key > row,
// i.e. c > r (first 16 keys) / c + 16 > r (second 16 keys).
// ---------------------------------------------------------------------------
template <bool MASKED>
__device__ __forceinline__ void softmax_update(
    const v8f& s0v, const v8f& s1v, float* alpha8,
    float* mrow8, float* lrow8,
    __bf16 (*Ps)[40], int prow, int ln, int c)
{
#pragma unroll
    for (int r = 0; r < 8; ++r) {
        float s0 = s0v[r];
        float s1 = s1v[r];
        if (MASKED) {
            if (c > r)      s0 = -3.0e38f;
            if (c + 16 > r) s1 = -3.0e38f;
        }
        float rm = fmaxf(s0, s1);
        rm = fmaxf(rm, __shfl_xor(rm, 1, 32));
        rm = fmaxf(rm, __shfl_xor(rm, 2, 32));
        rm = fmaxf(rm, __shfl_xor(rm, 4, 32));
        rm = fmaxf(rm, __shfl_xor(rm, 8, 32));
        const float mnew = fmaxf(mrow8[r], rm);
        const float a  = __expf(mrow8[r] - mnew);
        const float p0 = __expf(s0 - mnew);
        const float p1 = __expf(s1 - mnew);
        float ps = p0 + p1;
        ps += __shfl_xor(ps, 1, 32);
        ps += __shfl_xor(ps, 2, 32);
        ps += __shfl_xor(ps, 4, 32);
        ps += __shfl_xor(ps, 8, 32);
        lrow8[r] = lrow8[r] * a + ps;
        mrow8[r] = mnew;
        alpha8[r] = a;
        Ps[prow + r][ln]      = (__bf16)p0;
        Ps[prow + r][16 + ln] = (__bf16)p1;
    }
}

// ---------------------------------------------------------------------------
// One 32-key attention step for a 32-row query tile:
// 8 WMMA (QK^T over HD=64) + softmax + 8 WMMA (PV into 32x64 f32 acc).
// ---------------------------------------------------------------------------
template <bool MASKED>
__device__ __forceinline__ void attn_tile_step(
    int kbase, int q0, int lg, int ln,
    const __bf16* __restrict__ Kbase, const __bf16* __restrict__ Vbase,
    __bf16 (*Qs)[72], __bf16 (*Ps)[40],
    v8f (&oacc)[2][4], float (&mrow)[16], float (&lrow)[16])
{
    // K fragments (B operand), shared by both M tiles
    Frag16 kf[2][2];
#pragma unroll
    for (int nt = 0; nt < 2; ++nt)
#pragma unroll
        for (int dk = 0; dk < 2; ++dk) {
            const __bf16* p = Kbase + (size_t)(kbase + nt * 16 + ln) * HD + dk * 32 + lg * 16;
            kf[nt][dk].q[0] = *(const uint4*)p;
            kf[nt][dk].q[1] = *(const uint4*)(p + 8);
        }

    v8f zero = {};
#pragma unroll
    for (int mi = 0; mi < 2; ++mi) {
        // transient Q fragments for this M tile, straight from LDS
        Frag16 qf0, qf1;
        const __bf16* p0 = &Qs[mi * 16 + ln][lg * 16];
        qf0.q[0] = *(const uint4*)p0;
        qf0.q[1] = *(const uint4*)(p0 + 8);
        const __bf16* p1 = &Qs[mi * 16 + ln][32 + lg * 16];
        qf1.q[0] = *(const uint4*)p1;
        qf1.q[1] = *(const uint4*)(p1 + 8);

        v8f sc[2] = {zero, zero};
#pragma unroll
        for (int nt = 0; nt < 2; ++nt) {
            sc[nt] = wmma_bf16(qf0.v, kf[nt][0].v, sc[nt]);
            sc[nt] = wmma_bf16(qf1.v, kf[nt][1].v, sc[nt]);
        }
        const int prow = mi * 16 + lg * 8;
        float alpha[8];
        softmax_update<MASKED>(sc[0], sc[1], alpha, mrow + mi * 8, lrow + mi * 8,
                               Ps, prow, ln, kbase + ln - q0 - prow);
        // accumulator rescale on one code path, outside any masking logic
#pragma unroll
        for (int ni = 0; ni < 4; ++ni)
#pragma unroll
            for (int r = 0; r < 8; ++r) oacc[mi][ni][r] *= alpha[r];
    }

    __syncthreads();  // single-wave WG: S_NOP, but orders LDS store->load

    // P as A operand (one 16x32 fragment per M tile)
    Frag16 pf[2];
#pragma unroll
    for (int mi = 0; mi < 2; ++mi) {
        const __bf16* pp = &Ps[mi * 16 + ln][lg * 16];
        pf[mi].q[0] = *(const uint4*)pp;
        pf[mi].q[1] = *(const uint4*)(pp + 8);
    }
    // V fragments (B operand): each fetched once, used by both M tiles
#pragma unroll
    for (int ni = 0; ni < 4; ++ni) {
        Frag16 vf;
        const __bf16* vp = Vbase + (size_t)(ni * 16 + ln) * Ss + kbase + lg * 16;
        vf.q[0] = *(const uint4*)vp;
        vf.q[1] = *(const uint4*)(vp + 8);
#pragma unroll
        for (int mi = 0; mi < 2; ++mi)
            oacc[mi][ni] = wmma_bf16(pf[mi].v, vf.v, oacc[mi][ni]);
    }
    __syncthreads();
}

// ---------------------------------------------------------------------------
// Flash attention: one wave32 per 32 query rows (two 16-row M tiles share all
// K/V fragment fetches). Q staged in LDS keeps the kernel well under 256 VGPRs.
// The single causally-partial key tile is PEELED out of the main loop, so the
// hot loop is one straight-line branch-free path (no live-range splits / acc
// copies at control-flow joins). Q arrives pre-scaled by 1/sqrt(HD).
// ---------------------------------------------------------------------------
__global__ __launch_bounds__(32) void flash_attn(
    const __bf16* __restrict__ Qb,   // [B*H][S][HD] (pre-scaled)
    const __bf16* __restrict__ Kb,   // [B*H][S][HD]
    const __bf16* __restrict__ Vt,   // [B*H][HD][S]
    __bf16* __restrict__ Ob)         // [B][S][D]
{
    __shared__ __bf16 Qs[32][72];    // Q tile (stride 72 -> conflict-free b128)
    __shared__ __bf16 Ps[32][40];    // P tile bounce buffer (C-layout -> A-layout)

    const int lane = threadIdx.x & 31;
    const int lg = lane >> 4, ln = lane & 15;
    const int qt = blockIdx.x & 63;       // S/32 = 64 query tiles
    const int bh = blockIdx.x >> 6;       // 0..31
    const int b = bh >> 4, h = bh & 15;
    const int q0 = qt * 32;

    const __bf16* Qbase = Qb + (size_t)bh * Ss * HD;
    const __bf16* Kbase = Kb + (size_t)bh * Ss * HD;
    const __bf16* Vbase = Vt + (size_t)bh * HD * Ss;

    // stage the 32x64 Q tile into LDS (8 x 16B chunks per lane)
#pragma unroll
    for (int i = 0; i < 8; ++i) {
        const int t = lane + i * 32;        // 0..255
        const int row = t >> 3, seg = t & 7;
        *(uint4*)&Qs[row][seg * 8] =
            *(const uint4*)(Qbase + (size_t)(q0 + row) * HD + seg * 8);
    }
    __syncthreads();

    v8f oacc[2][4];
    v8f zero = {};
#pragma unroll
    for (int mi = 0; mi < 2; ++mi)
#pragma unroll
        for (int ni = 0; ni < 4; ++ni) oacc[mi][ni] = zero;
    float mrow[16], lrow[16];
#pragma unroll
    for (int r = 0; r < 16; ++r) { mrow[r] = -3.0e38f; lrow[r] = 0.0f; }

    // hot loop: causally interior tiles, branch-free body
    for (int kt = 0; kt < qt; ++kt)
        attn_tile_step<false>(kt * 32, q0, lg, ln, Kbase, Vbase, Qs, Ps,
                              oacc, mrow, lrow);
    // peeled: the one partial tile on the causal diagonal
    attn_tile_step<true>(q0, q0, lg, ln, Kbase, Vbase, Qs, Ps,
                         oacc, mrow, lrow);

    // normalize (one reciprocal per row) and write O[b][s][h*64+d]
#pragma unroll
    for (int mi = 0; mi < 2; ++mi)
#pragma unroll
        for (int r = 0; r < 8; ++r) {
            const float inv = 1.0f / lrow[mi * 8 + r];
            const int qrow = q0 + mi * 16 + r + lg * 8;
            __bf16* orow = Ob + ((size_t)b * Ss + qrow) * Dd + h * HD + ln;
#pragma unroll
            for (int ni = 0; ni < 4; ++ni)
                orow[ni * 16] = (__bf16)(oacc[mi][ni][r] * inv);
        }
}

// ---------------------------------------------------------------------------
// Launch
// ---------------------------------------------------------------------------
extern "C" void kernel_launch(void* const* d_in, const int* in_sizes, int n_in,
                              void* d_out, int out_size, void* d_ws, size_t ws_size,
                              hipStream_t stream) {
    const float* hs   = (const float*)d_in[0];  // [2,2048,1024]
    const float* Wat  = (const float*)d_in[1];  // [1024,3072]
    const float* batt = (const float*)d_in[2];  // [3072]
    const float* Wpr  = (const float*)d_in[3];  // [1024,1024]
    const float* bpr  = (const float*)d_in[4];  // [1024]
    float* out = (float*)d_out;                 // [2,2048,1024] f32

    char* ws = (char*)d_ws;
    const size_t MB = 1u << 20;
    __bf16* Xb  = (__bf16*)(ws);             // 8 MB   X bf16
    __bf16* Wab = (__bf16*)(ws + 8 * MB);    // 6 MB   W_attn bf16
    __bf16* Wpb = (__bf16*)(ws + 14 * MB);   // 2 MB   W_proj bf16
    __bf16* Qb  = (__bf16*)(ws + 16 * MB);   // 8 MB   Q [B,H,S,HD] (pre-scaled)
    __bf16* Kb  = (__bf16*)(ws + 24 * MB);   // 8 MB   K [B,H,S,HD]
    __bf16* Vt  = (__bf16*)(ws + 32 * MB);   // 8 MB   V^T [B,H,HD,S]
    __bf16* Ob  = (__bf16*)(ws + 40 * MB);   // 8 MB   attn out [B,S,D]

    cvt_f32_bf16<<<(Mtot * Ktot) / 1024, 256, 0, stream>>>(hs, Xb);
    cvt_f32_bf16<<<(Ktot * 3 * Dd) / 1024, 256, 0, stream>>>(Wat, Wab);
    cvt_f32_bf16<<<(Ktot * Dd) / 1024, 256, 0, stream>>>(Wpr, Wpb);

    gemm_bf16<0, 3 * Dd><<<dim3((3 * Dd) / 128, Mtot / 128), 256, 0, stream>>>(
        Xb, Wab, batt, Qb, Kb, Vt, nullptr);

    flash_attn<<<Bb * Hh * (Ss / 32), 32, 0, stream>>>(Qb, Kb, Vt, Ob);

    gemm_bf16<1, Dd><<<dim3(Dd / 128, Mtot / 128), 256, 0, stream>>>(
        Ob, Wpb, bpr, nullptr, nullptr, nullptr, out);
}
